// MessagePassing_13872744366649
// MI455X (gfx1250) — compile-verified, matrix operand order checked by reading, not run on hardware
//
#include <hip/hip_runtime.h>
#include <hip/hip_bf16.h>
#include <stdint.h>

// ---------------------------------------------------------------------------
// GIN message-passing network for MI455X (gfx1250, wave32, WMMA).
//  - edge segment-sum: vectorized gather + f32 L2 atomics (data resident in L2)
//  - all GEMMs: bf16 V_WMMA_F32_16X16X32_BF16, f32 accum, bias+ReLU epilogue
//  - weights repacked once into WMMA B-fragment lane order (contiguous b128 loads)
//  - M % 16 == 0 at every call site -> wave-uniform early exit, no exec masking
//    anywhere in the WMMA loop or epilogue
// ---------------------------------------------------------------------------

typedef __attribute__((ext_vector_type(16))) __bf16 v16bf;
typedef __attribute__((ext_vector_type(8)))  float  v8f;

#define NNODES 50000
#define NEDGES 800000
#define NFEAT  9
#define EMB    128
#define HID    256
#define NLAY   3
#define NGRAPH 256

// -------------------------- small elementwise kernels ----------------------

__global__ __launch_bounds__(256) void k_pad(const float* __restrict__ x,
                                             float* __restrict__ h, int n) {
  int t = blockIdx.x * 256 + threadIdx.x;
  if (t >= n * EMB) return;
  int node = t >> 7, f = t & 127;
  h[t] = (f < NFEAT) ? x[node * NFEAT + f] : 0.0f;
}

__global__ __launch_bounds__(256) void k_copy4(const float4* __restrict__ src,
                                               float4* __restrict__ dst, int n4) {
  int t = blockIdx.x * 256 + threadIdx.x;
  if (t < n4) dst[t] = src[t];
}

__global__ __launch_bounds__(256) void k_zero(float* __restrict__ p, int n) {
  int t = blockIdx.x * 256 + threadIdx.x;
  if (t < n) p[t] = 0.0f;
}

// -------------------------- edge scatter (segment_sum) ---------------------
// 32 lanes per edge, float4 per lane: 128-feature row gathered with b128 loads,
// accumulated into agg[dst] with global_atomic_add_f32 (L2-resident data).

__global__ __launch_bounds__(256) void k_edge_scatter(
    const long long* __restrict__ src, const long long* __restrict__ dst,
    const float* __restrict__ h, float* __restrict__ agg, int e) {
  int t = blockIdx.x * 256 + threadIdx.x;
  int edge = t >> 5;
  if (edge >= e) return;
  int q = (t & 31) << 2;
  long long s = src[edge];
  long long d = dst[edge];
  float4 v = *(const float4*)(h + (size_t)s * EMB + q);
  float* ap = agg + (size_t)d * EMB + q;
  atomicAdd(ap + 0, v.x);
  atomicAdd(ap + 1, v.y);
  atomicAdd(ap + 2, v.z);
  atomicAdd(ap + 3, v.w);
}

// -------------------------- weight repack to B-fragment order --------------
// Packed layout: frag (kb,nb) -> 32 lanes * 16 halves contiguous.
// Lane L holds column n = nb*16 + (L&15); element i holds
//   k_local = (i>=8 ? 16 : 0) + (L>>4)*8 + (i&7)     (two contiguous runs of 8)

__global__ __launch_bounds__(256) void k_repack_w(const float* __restrict__ W,
                                                  __bf16* __restrict__ out,
                                                  int K, int NOUT) {
  int t = blockIdx.x * 256 + threadIdx.x;
  if (t >= K * NOUT) return;
  int i    = t & 15;
  int lane = (t >> 4) & 31;
  int tile = t >> 9;
  int NT = NOUT >> 4;
  int nb = tile % NT;
  int kb = tile / NT;
  int n  = nb * 16 + (lane & 15);
  int kl = ((i >= 8) ? 16 : 0) + ((lane >> 4) << 3) + (i & 7);
  int k  = (kb << 5) + kl;
  out[t] = (__bf16)W[(size_t)k * NOUT + n];
}

// -------------------------- WMMA GEMM + bias + optional ReLU ---------------
// Block = 4 waves (128 thr), 64 rows per block, 16 rows per wave.
// A fragments kept resident in VGPRs; B fragments are contiguous b128 loads.
// Requires M % 16 == 0 (true for all call sites: N=50000, G=256).

template <int K, int NOUT, bool A_F32, bool RELU, bool OUT_BF16>
__global__ __launch_bounds__(128) void k_gemm(
    const void* __restrict__ Av, const __bf16* __restrict__ Bp,
    const float* __restrict__ bias, void* __restrict__ Outv, int M) {
  constexpr int KT = K / 32;
  constexpr int NT = NOUT / 16;
  const int wave = threadIdx.x >> 5;
  const int lane = threadIdx.x & 31;
  const int row0 = blockIdx.x * 64 + wave * 16;
  if (row0 >= M) return;  // wave-uniform: whole wave exits, no WMMA executed
  const int mrow = lane & 15;
  const int hlf  = lane >> 4;
  const int r = row0 + mrow;   // always < M since M % 16 == 0

  // ---- load A fragments (held resident across the whole N loop) ----
  // Lane layout: two contiguous 8-element runs at k = kb*32 + hlf*8 (+16).
  v16bf afrag[KT];
  if (A_F32) {
    const float* rowp = (const float*)Av + (size_t)r * K;
#pragma unroll
    for (int kb = 0; kb < KT; ++kb) {
      const float* p0 = rowp + kb * 32 + hlf * 8;
      float4 f0 = *(const float4*)(p0);
      float4 f1 = *(const float4*)(p0 + 4);
      float4 f2 = *(const float4*)(p0 + 16);
      float4 f3 = *(const float4*)(p0 + 20);
      v16bf a;
      a[0]  = (__bf16)f0.x; a[1]  = (__bf16)f0.y;
      a[2]  = (__bf16)f0.z; a[3]  = (__bf16)f0.w;
      a[4]  = (__bf16)f1.x; a[5]  = (__bf16)f1.y;
      a[6]  = (__bf16)f1.z; a[7]  = (__bf16)f1.w;
      a[8]  = (__bf16)f2.x; a[9]  = (__bf16)f2.y;
      a[10] = (__bf16)f2.z; a[11] = (__bf16)f2.w;
      a[12] = (__bf16)f3.x; a[13] = (__bf16)f3.y;
      a[14] = (__bf16)f3.z; a[15] = (__bf16)f3.w;
      afrag[kb] = a;
    }
  } else {
    const __bf16* rowp = (const __bf16*)Av + (size_t)r * K;
#pragma unroll
    for (int kb = 0; kb < KT; ++kb) {
      const __bf16* p0 = rowp + kb * 32 + hlf * 8;
      union { v16bf v; uint4 q[2]; } u;
      u.q[0] = *(const uint4*)(p0);        // 8 bf16 = 16 bytes (b128)
      u.q[1] = *(const uint4*)(p0 + 16);   // 8 bf16 = 16 bytes (b128)
      afrag[kb] = u.v;
    }
  }

  // ---- loop over 16-wide output column tiles ----
  for (int nb = 0; nb < NT; ++nb) {
    v8f acc = {};
#pragma unroll
    for (int kb = 0; kb < KT; ++kb) {
      v16bf b = *(const v16bf*)(Bp + ((size_t)(kb * NT + nb) * 32 + lane) * 16);
      acc = __builtin_amdgcn_wmma_f32_16x16x32_bf16(
          false, afrag[kb], false, b, (short)0, acc, false, false);
    }
    const int col = nb * 16 + (lane & 15);
    const float bv = bias[col];
    // C layout: lane -> column (lane&15); VGPR j -> row j + 8*(lane>>4)
#pragma unroll
    for (int j = 0; j < 8; ++j) {
      int orow = row0 + j + 8 * hlf;
      float v = acc[j] + bv;
      if (RELU) v = v > 0.0f ? v : 0.0f;
      if (OUT_BF16)
        ((__bf16*)Outv)[(size_t)orow * NOUT + col] = (__bf16)v;
      else
        ((float*)Outv)[(size_t)orow * NOUT + col] = v;
    }
  }
}

// -------------------------- pooling --------------------------------------

__global__ __launch_bounds__(256) void k_pool_acc(
    const float* __restrict__ h, const long long* __restrict__ batch,
    float* __restrict__ sums, float* __restrict__ cnts, int n) {
  int t = blockIdx.x * 256 + threadIdx.x;
  int node = t >> 5;
  if (node >= n) return;
  int q = (t & 31) << 2;
  long long g = batch[node];
  float4 v = *(const float4*)(h + (size_t)node * EMB + q);
  float* sp = sums + (size_t)g * EMB + q;
  atomicAdd(sp + 0, v.x);
  atomicAdd(sp + 1, v.y);
  atomicAdd(sp + 2, v.z);
  atomicAdd(sp + 3, v.w);
  if ((t & 31) == 0) atomicAdd(cnts + g, 1.0f);
}

__global__ __launch_bounds__(256) void k_pool_finalize(
    const float* __restrict__ sums, const float* __restrict__ cnts,
    __bf16* __restrict__ pooled) {
  int t = blockIdx.x * 256 + threadIdx.x;
  if (t >= NGRAPH * EMB) return;
  int g = t >> 7, f = t & 127;
  float s = sums[t];
  float c = cnts[g];
  if (c < 1.0f) c = 1.0f;
  pooled[(size_t)g * (2 * EMB) + f]       = (__bf16)s;
  pooled[(size_t)g * (2 * EMB) + EMB + f] = (__bf16)(s / c);
}

// -------------------------- host launcher ---------------------------------

extern "C" void kernel_launch(void* const* d_in, const int* in_sizes, int n_in,
                              void* d_out, int out_size, void* d_ws, size_t ws_size,
                              hipStream_t stream) {
  const float* x       = (const float*)d_in[0];
  const long long* ei  = (const long long*)d_in[1];   // int64 per reference
  const long long* bat = (const long long*)d_in[2];
  const float* gin_w1  = (const float*)d_in[3];
  const float* gin_b1  = (const float*)d_in[4];
  const float* gin_w2  = (const float*)d_in[5];
  const float* gin_b2  = (const float*)d_in[6];
  const float* post_w1 = (const float*)d_in[7];
  const float* post_b1 = (const float*)d_in[8];
  const float* post_w2 = (const float*)d_in[9];
  const float* post_b2 = (const float*)d_in[10];
  float* out = (float*)d_out;

  const int N = in_sizes[0] / NFEAT;
  const int E = in_sizes[1] / 2;
  const long long* esrc = ei;
  const long long* edst = ei + E;

  // workspace carve-out (256B aligned slices)
  char* wp = (char*)d_ws;
  auto carve = [&](size_t bytes) {
    void* r = (void*)wp;
    wp += (bytes + 255) & ~(size_t)255;
    return r;
  };
  float*  h      = (float*)carve((size_t)N * EMB * 4);
  float*  agg    = (float*)carve((size_t)N * EMB * 4);
  __bf16* z      = (__bf16*)carve((size_t)N * HID * 2);
  __bf16* wp1    = (__bf16*)carve((size_t)NLAY * EMB * HID * 2);
  __bf16* wp2    = (__bf16*)carve((size_t)NLAY * HID * EMB * 2);
  __bf16* pwp1   = (__bf16*)carve((size_t)(2 * EMB) * HID * 2);
  __bf16* pwp2   = (__bf16*)carve((size_t)HID * EMB * 2);
  float*  sums   = (float*)carve((size_t)NGRAPH * EMB * 4);
  float*  cnts   = (float*)carve((size_t)NGRAPH * 4);
  __bf16* pooled = (__bf16*)carve((size_t)NGRAPH * 2 * EMB * 2);
  __bf16* hidden = (__bf16*)carve((size_t)NGRAPH * HID * 2);
  (void)ws_size; (void)n_in; (void)out_size;

  // 1) repack all weights into WMMA B-fragment order (bf16)
  for (int l = 0; l < NLAY; ++l) {
    k_repack_w<<<(EMB * HID + 255) / 256, 256, 0, stream>>>(
        gin_w1 + (size_t)l * EMB * HID, wp1 + (size_t)l * EMB * HID, EMB, HID);
    k_repack_w<<<(HID * EMB + 255) / 256, 256, 0, stream>>>(
        gin_w2 + (size_t)l * HID * EMB, wp2 + (size_t)l * HID * EMB, HID, EMB);
  }
  k_repack_w<<<(2 * EMB * HID + 255) / 256, 256, 0, stream>>>(post_w1, pwp1, 2 * EMB, HID);
  k_repack_w<<<(HID * EMB + 255) / 256, 256, 0, stream>>>(post_w2, pwp2, HID, EMB);

  // 2) pad raw features into h [N,128]
  k_pad<<<((size_t)N * EMB + 255) / 256, 256, 0, stream>>>(x, h, N);

  // 3) six GIN conv steps
  const int gemm_blocks = (N + 63) / 64;
  for (int it = 0; it < 2 * NLAY; ++it) {
    const int l = it >> 1;
    // agg = h  (GIN eps=0: z = h + sum_neighbors)
    k_copy4<<<((size_t)N * (EMB / 4) + 255) / 256, 256, 0, stream>>>(
        (const float4*)h, (float4*)agg, N * (EMB / 4));
    // agg[dst] += h[src]
    k_edge_scatter<<<((size_t)E * 32 + 255) / 256, 256, 0, stream>>>(
        esrc, edst, h, agg, E);
    // z = relu(agg @ w1 + b1)   [N,128]x[128,256] -> bf16
    k_gemm<EMB, HID, true, true, true><<<gemm_blocks, 128, 0, stream>>>(
        agg, wp1 + (size_t)l * EMB * HID, gin_b1 + (size_t)l * HID, z, N);
    // h = relu(z @ w2 + b2)     [N,256]x[256,128] -> f32
    k_gemm<HID, EMB, false, true, false><<<gemm_blocks, 128, 0, stream>>>(
        z, wp2 + (size_t)l * HID * EMB, gin_b2 + (size_t)l * EMB, h, N);
  }

  // 4) global add / mean pooling
  k_zero<<<(NGRAPH * (EMB + 1) + 255) / 256, 256, 0, stream>>>(sums, NGRAPH * EMB);
  k_zero<<<1, 256, 0, stream>>>(cnts, NGRAPH);
  k_pool_acc<<<((size_t)N * 32 + 255) / 256, 256, 0, stream>>>(h, bat, sums, cnts, N);
  k_pool_finalize<<<(NGRAPH * EMB + 255) / 256, 256, 0, stream>>>(sums, cnts, pooled);

  // 5) post MLP: [256,256]x[256,256] relu -> [256,256]x[256,128]
  const int post_blocks = (NGRAPH + 63) / 64;
  k_gemm<2 * EMB, HID, false, true, true><<<post_blocks, 128, 0, stream>>>(
      pooled, pwp1, post_b1, hidden, NGRAPH);
  k_gemm<HID, EMB, false, false, false><<<post_blocks, 128, 0, stream>>>(
      hidden, pwp2, post_b2, out, NGRAPH);
}
